// GuidedCxtAtten_8615704396289
// MI455X (gfx1250) — compile-verified, matrix operand order checked by reading, not run on hardware
//
#include <hip/hip_runtime.h>
#include <cstdint>
#include <cstddef>

// ---- problem constants (from reference setup_inputs) ----
#define B_   4
#define CG   256
#define CO   128      // Cg/2
#define CA   128
#define H_   128
#define W_   128
#define HS   64
#define L_   4096     // HS*WS
#define K1   1152     // CO*3*3
#define EV   2048     // CA*4*4

// WMMA types (gfx1250, wave32): bf16 16x16x32, f32 accum
typedef __attribute__((ext_vector_type(16))) __bf16 v16bf;
typedef __attribute__((ext_vector_type(8)))  float  f32x8;
typedef __attribute__((ext_vector_type(4)))  unsigned int u32x4;
typedef __attribute__((ext_vector_type(4)))  int i32x4;
typedef __attribute__((ext_vector_type(8)))  int i32x8;
union FragBF { unsigned int u[8]; v16bf v; };

__device__ __forceinline__ unsigned short f2bf(float f) {
  union { float f; unsigned int u; } x; x.f = f;
  unsigned int u = x.u;
  u += 0x7FFFu + ((u >> 16) & 1u);   // round-to-nearest-even
  return (unsigned short)(u >> 16);
}
__device__ __forceinline__ int refl64(int t)  { return t < 0 ? -t : (t > 63  ? 126 - t : t); }
__device__ __forceinline__ int refl128(int t) { return t < 0 ? -t : (t > 127 ? 254 - t : t); }

// --- Tensor Data Mover: 2D tile (2-byte elements) global -> LDS (ISA cdna5 ch7/8) ---
// D# group0: count=1 | lds_addr | global_addr[56:0] | type=2
// D# group1: data_size=1(2B) | tensor_dim0/1 | tile_dim0/1 (tile_dim2=0 -> 2D) | dim0 stride
__device__ __forceinline__ void tdm_load_tile_2d(unsigned lds_off, unsigned long long gaddr,
                                                 unsigned tensor_d0, unsigned tensor_d1,
                                                 unsigned tile_d0, unsigned tile_d1,
                                                 unsigned stride_d0) {
  u32x4 g0;
  g0[0] = 1u;                                                       // count=1 (valid user D#)
  g0[1] = lds_off;                                                  // lds_addr (bytes)
  g0[2] = (unsigned)(gaddr & 0xFFFFFFFFull);                        // global_addr[31:0]
  g0[3] = (unsigned)((gaddr >> 32) & 0x01FFFFFFull) | 0x80000000u;  // addr[56:32] | type=2
  i32x8 g1;
  g1[0] = (int)(1u << 16);                                          // data_size=1 -> 2 bytes
  g1[1] = (int)((tensor_d0 & 0xFFFFu) << 16);                       // tensor_dim0[15:0]
  g1[2] = (int)((tensor_d0 >> 16) | ((tensor_d1 & 0xFFFFu) << 16)); // dim0 hi | dim1 lo
  g1[3] = (int)((tensor_d1 >> 16) | (tile_d0 << 16));               // dim1 hi | tile_dim0
  g1[4] = (int)(tile_d1 & 0xFFFFu);                                 // tile_dim1 (tile_dim2=0)
  g1[5] = (int)stride_d0;                                           // tensor_dim0_stride lo
  g1[6] = 0; g1[7] = 0;
  i32x4 g2 = {0, 0, 0, 0};
  i32x4 g3 = {0, 0, 0, 0};
#if defined(__clang_major__) && (__clang_major__ >= 23)
  i32x8 g4 = {0, 0, 0, 0, 0, 0, 0, 0};
  __builtin_amdgcn_tensor_load_to_lds(g0, g1, g2, g3, g4, 0);
#else
  __builtin_amdgcn_tensor_load_to_lds(g0, g1, g2, g3, 0);
#endif
}

// ---------------------------------------------------------------- init
__global__ void k_init(float* y, float* bnsum, float* bnsq) {
  size_t idx = (size_t)blockIdx.x * 256 + threadIdx.x;
  if (idx < (size_t)B_ * CA * H_ * W_) y[idx] = 0.f;
  if (idx < CA) { bnsum[idx] = 0.f; bnsq[idx] = 0.f; }
}

// ---------------------------------------------------------------- su/sk + softmax_scale output
__global__ void k_meta(const float* __restrict__ unk, float* susk, float* out) {
  const int b = blockIdx.x, tid = threadIdx.x;
  __shared__ float red[256];
  float s = 0.f;
  for (int i = tid; i < L_; i += 256) {
    int py = i >> 6, px = i & 63;
    s += unk[(size_t)b * H_ * W_ + (2 * py) * W_ + 2 * px];
  }
  red[tid] = s; __syncthreads();
  for (int st = 128; st > 0; st >>= 1) { if (tid < st) red[tid] += red[tid + st]; __syncthreads(); }
  if (tid == 0) {
    float u = red[0] / (float)L_;
    float k = 1.0f - u;
    float su = sqrtf(u / fmaxf(k, 1e-20f)); su = fminf(fmaxf(su, 0.1f), 10.0f);
    float sk = sqrtf(k / fmaxf(u, 1e-20f)); sk = fminf(fmaxf(sk, 0.1f), 10.0f);
    susk[2 * b] = su; susk[2 * b + 1] = sk;
    out[(size_t)B_ * CA * H_ * W_ + B_ * 2 * HS * HS + 2 * b]     = su;
    out[(size_t)B_ * CA * H_ * W_ + B_ * 2 * HS * HS + 2 * b + 1] = sk;
  }
}

// ---------------------------------------------------------------- fds = (gc_w @ f + gc_b) at even pixels
__global__ void k_fds(const float* __restrict__ f, const float* __restrict__ gc_w,
                      const float* __restrict__ gc_b, float* __restrict__ fds) {
  int idx = blockIdx.x * 256 + threadIdx.x;
  int p = idx & (L_ - 1), o = (idx >> 12) & 127, b = idx >> 19;
  int py = p >> 6, px = p & 63;
  const float* fb = f + (size_t)b * CG * H_ * W_ + (size_t)(2 * py) * W_ + 2 * px;
  const float* wr = gc_w + o * CG;
  float acc = gc_b[o];
  for (int i = 0; i < CG; ++i) acc += wr[i] * fb[(size_t)i * H_ * W_];
  fds[((size_t)(b * CO + o)) * L_ + p] = acc;
}

// ---------------------------------------------------------------- P (bf16 3x3 patches), norms->rowscale, mm
__global__ void k_pack_P(const float* __restrict__ fds, const float* __restrict__ unk,
                         const float* __restrict__ susk,
                         unsigned short* __restrict__ P, float* __restrict__ rs,
                         float* __restrict__ mmv) {
  const int q = blockIdx.x & (L_ - 1), b = blockIdx.x >> 12;
  const int c = threadIdx.x;               // 0..127 (channel)
  const int py = q >> 6, px = q & 63;
  float ss = 0.f;
  unsigned short* Pr = P + ((size_t)(b * L_ + q)) * K1 + c * 9;
  const float* fc = fds + ((size_t)(b * CO + c)) * L_;
  #pragma unroll
  for (int di = 0; di < 3; ++di) {
    int ry = refl64(py + di - 1);
    #pragma unroll
    for (int dj = 0; dj < 3; ++dj) {
      int rx = refl64(px + dj - 1);
      float v = fc[ry * 64 + rx];
      Pr[di * 3 + dj] = f2bf(v);
      ss += v * v;
    }
  }
  __shared__ float red[128];
  red[c] = ss; __syncthreads();
  for (int st = 64; st > 0; st >>= 1) { if (c < st) red[c] += red[c + st]; __syncthreads(); }
  if (c == 0) {
    float norm = sqrtf(red[0]);
    float m = 0.f;
    const float* ub = unk + (size_t)b * H_ * W_;
    for (int di = 0; di < 3; ++di) {
      int ry = refl64(py + di - 1);
      for (int dj = 0; dj < 3; ++dj) {
        int rx = refl64(px + dj - 1);
        m += ub[(2 * ry) * W_ + 2 * rx];
      }
    }
    float mm = (m * (1.0f / 9.0f) > 0.f) ? 1.f : 0.f;
    float gate = (mm > 0.f) ? susk[2 * b] : susk[2 * b + 1];
    rs[b * L_ + q]  = gate / fmaxf(norm, 1e-4f);
    mmv[b * L_ + q] = mm;
  }
}

// ---------------------------------------------------------------- V (bf16 alpha 4x4 patches, kernel=4 stride=2, reflect pad 1)
__global__ void k_pack_V(const float* __restrict__ alpha, unsigned short* __restrict__ V) {
  const int q = blockIdx.x & (L_ - 1), b = blockIdx.x >> 12;
  const int py = q >> 6, px = q & 63;
  unsigned short* Vr = V + ((size_t)(b * L_ + q)) * EV;
  for (int e = threadIdx.x; e < EV; e += 128) {
    int c = e >> 4, di = (e >> 2) & 3, dj = e & 3;
    int h = refl128(2 * py + di - 1);
    int w = refl128(2 * px + dj - 1);
    Vr[e] = f2bf(alpha[(((size_t)b * CA + c) * H_ + h) * W_ + w]);
  }
}

// ---------------------------------------------------------------- GEMM1: scores[b][q][p] = rs[q]*(P[q]·P[p]) - 1e4*mm[q]*(q==p)
// WG tile 128q x 128p, wave = 64q x 32p (8 WMMAs / wave / k-step).
// A/B tiles staged by the Tensor Data Mover (one wave issues, TENSORcnt waited, WG barrier).
__global__ __launch_bounds__(256)
void k_gemm1(const unsigned short* __restrict__ P, const float* __restrict__ rs,
             const float* __restrict__ mmv, float* __restrict__ scores) {
  const int b  = blockIdx.z;
  const int p0 = blockIdx.x * 128;
  const int q0 = blockIdx.y * 128;
  const int tid = threadIdx.x;
  const int lane = tid & 31, wid = tid >> 5;
  const int qw = wid >> 2, pw = wid & 3;       // wave -> 64 rows (q) x 32 cols (p)
  const int ln = lane & 15, hi = lane >> 4;

  __shared__ unsigned int As[128 * 16];        // [q row][k u32-pair]  (128 x 64B rows)
  __shared__ unsigned int Bs[128 * 16];        // [p row][k u32-pair]

  const unsigned long long gA = (unsigned long long)(uintptr_t)(P + (size_t)(b * L_ + q0) * K1);
  const unsigned long long gB = (unsigned long long)(uintptr_t)(P + (size_t)(b * L_ + p0) * K1);
  const unsigned ldsA = (unsigned)(uintptr_t)(void*)As;
  const unsigned ldsB = (unsigned)(uintptr_t)(void*)Bs;

  const f32x8 z = {0.f,0.f,0.f,0.f,0.f,0.f,0.f,0.f};
  f32x8 acc[4][2] = {{z, z}, {z, z}, {z, z}, {z, z}};

  for (int ks = 0; ks < K1 / 32; ++ks) {
    __syncthreads();                           // previous iteration's LDS reads done
    if (tid < 32) {                            // wave 0 drives the TDM
      tdm_load_tile_2d(ldsA, gA + (unsigned long long)ks * 64ull,
                       K1, 128, 32, 128, K1);
      tdm_load_tile_2d(ldsB, gB + (unsigned long long)ks * 64ull,
                       K1, 128, 32, 128, K1);
      __builtin_amdgcn_s_wait_tensorcnt(0);
    }
    __syncthreads();                           // tiles visible to all waves

    FragBF fb[2];
    #pragma unroll
    for (int tj = 0; tj < 2; ++tj) {
      const int br = pw * 32 + tj * 16 + ln;   // B: lane ln holds col n=ln
      #pragma unroll
      for (int v = 0; v < 8; ++v)
        fb[tj].u[v] = Bs[br * 16 + ((v >> 2) * 8) + hi * 4 + (v & 3)];
    }
    #pragma unroll
    for (int ti = 0; ti < 4; ++ti) {
      FragBF a;
      const int ar = qw * 64 + ti * 16 + ln;   // A: lane ln holds row m=ln, hi selects k-half
      #pragma unroll
      for (int v = 0; v < 8; ++v)
        a.u[v] = As[ar * 16 + ((v >> 2) * 8) + hi * 4 + (v & 3)];
      #pragma unroll
      for (int tj = 0; tj < 2; ++tj)
        acc[ti][tj] = __builtin_amdgcn_wmma_f32_16x16x32_bf16(
            false, a.v, false, fb[tj].v, (short)0, acc[ti][tj], false, false);
    }
  }

  #pragma unroll
  for (int ti = 0; ti < 4; ++ti) {
    const int qb = q0 + qw * 64 + ti * 16 + hi * 8;   // C/D: m = v + 8*hi
    #pragma unroll
    for (int tj = 0; tj < 2; ++tj) {
      const int pc = p0 + pw * 32 + tj * 16 + ln;     // n = ln
      float* o = scores + ((size_t)(b * L_) + qb) * L_ + pc;
      #pragma unroll
      for (int v = 0; v < 8; ++v) {
        const int q = qb + v;
        float val = acc[ti][tj][v] * rs[b * L_ + q];
        if (q == pc) val -= 10000.0f * mmv[b * L_ + q];
        o[(size_t)v * L_] = val;
      }
    }
  }
}

// ---------------------------------------------------------------- column softmax stats + argmax -> offsets
__global__ void k_colreduce(const float* __restrict__ scores, float* __restrict__ cmax,
                            float* __restrict__ csum, float* __restrict__ out) {
  const int p = blockIdx.x * 256 + threadIdx.x;
  const int b = blockIdx.y;
  const float* col = scores + (size_t)b * L_ * L_ + p;
  float m = -3.4e38f; int am = 0;
  for (int q = 0; q < L_; ++q) { float s = col[(size_t)q * L_]; if (s > m) { m = s; am = q; } }
  float sum = 0.f;
  for (int q = 0; q < L_; ++q) sum += __expf(col[(size_t)q * L_] - m);
  cmax[b * L_ + p] = m;
  csum[b * L_ + p] = sum;
  const size_t base = (size_t)B_ * CA * H_ * W_;           // offsets region
  out[base + (size_t)b * 2 * L_ + p]      = (float)(am / HS - HS / 2);
  out[base + (size_t)b * 2 * L_ + L_ + p] = (float)(am % HS - HS / 2);
}

// ---------------------------------------------------------------- GEMM2: y += scatter( attn^T @ V ) / 4
__global__ __launch_bounds__(256)
void k_gemm2(const float* __restrict__ scores, const unsigned short* __restrict__ V,
             const float* __restrict__ cmax, const float* __restrict__ csum,
             float* __restrict__ y) {
  const int b  = blockIdx.z;
  const int e0 = blockIdx.x * 128;
  const int p0 = blockIdx.y * 64;
  const int tid = threadIdx.x;
  const int lane = tid & 31, wid = tid >> 5;
  const int ew = wid & 3, pw = wid >> 2;       // wave -> 32 p x 32 e
  const int ln = lane & 15, hi = lane >> 4;

  __shared__ unsigned int Pu[64 * 16];         // probs: [p][q u32-pair] bf16x2
  __shared__ unsigned int Vu[128 * 16];        // V^T:   [e][q u32-pair] bf16x2
  __shared__ float smax[64], srcp[64];
  if (tid < 64) {
    smax[tid] = cmax[b * L_ + p0 + tid];
    srcp[tid] = 1.0f / csum[b * L_ + p0 + tid];
  }

  const f32x8 z = {0.f,0.f,0.f,0.f,0.f,0.f,0.f,0.f};
  f32x8 acc[2][2] = {{z, z}, {z, z}};

  for (int ks = 0; ks < L_ / 32; ++ks) {
    const int q0 = ks * 32;
    __syncthreads();
    #pragma unroll
    for (int r = 0; r < 4; ++r) {              // probabilities -> bf16 pairs (transpose in LDS)
      int idx = tid + r * 256;                 // 0..1023
      int p = idx >> 4, j = idx & 15;
      const float* sp = scores + ((size_t)(b * L_) + q0 + 2 * j) * L_ + p0 + p;
      float pr0 = __expf(sp[0]          - smax[p]) * srcp[p];
      float pr1 = __expf(sp[(size_t)L_] - smax[p]) * srcp[p];
      Pu[idx] = (unsigned)f2bf(pr0) | ((unsigned)f2bf(pr1) << 16);
      if (ks + 1 < L_ / 32)                    // prefetch next scores slab
        __builtin_prefetch(sp + (size_t)32 * L_, 0, 0);
    }
    #pragma unroll
    for (int r = 0; r < 8; ++r) {              // V tile, transposed to [e][q-pair]
      int idx = tid + r * 256;                 // 0..2047
      int e = idx >> 4, j = idx & 15;
      const unsigned short* vp = V + ((size_t)(b * L_) + q0 + 2 * j) * EV + e0 + e;
      Vu[idx] = (unsigned)vp[0] | ((unsigned)vp[EV] << 16);
    }
    __syncthreads();
    #pragma unroll
    for (int ti = 0; ti < 2; ++ti) {
      FragBF a;
      const int ar = pw * 32 + ti * 16 + ln;   // A[m=p][k=q]
      #pragma unroll
      for (int v = 0; v < 8; ++v)
        a.u[v] = Pu[ar * 16 + ((v >> 2) * 8) + hi * 4 + (v & 3)];
      #pragma unroll
      for (int tj = 0; tj < 2; ++tj) {
        FragBF fb;
        const int br = ew * 32 + tj * 16 + ln; // B[k=q][n=e]
        #pragma unroll
        for (int v = 0; v < 8; ++v)
          fb.u[v] = Vu[br * 16 + ((v >> 2) * 8) + hi * 4 + (v & 3)];
        acc[ti][tj] = __builtin_amdgcn_wmma_f32_16x16x32_bf16(
            false, a.v, false, fb.v, (short)0, acc[ti][tj], false, false);
      }
    }
  }

  #pragma unroll
  for (int ti = 0; ti < 2; ++ti) {
    const int pb = p0 + pw * 32 + ti * 16 + hi * 8;     // m = p
    #pragma unroll
    for (int tj = 0; tj < 2; ++tj) {
      const int e = e0 + ew * 32 + tj * 16 + ln;        // n = e
      const int c = e >> 4, di = (e >> 2) & 3, dj = e & 3;
      #pragma unroll
      for (int v = 0; v < 8; ++v) {
        const int p = pb + v;
        const int py = p >> 6, px = p & 63;
        const int h = 2 * py + di - 1;
        const int w = 2 * px + dj - 1;
        if (h >= 0 && h < H_ && w >= 0 && w < W_)
          atomicAdd(&y[(((size_t)(b * CA) + c) * H_ + h) * W_ + w], acc[ti][tj][v] * 0.25f);
      }
    }
  }
}

// ---------------------------------------------------------------- BN stats over yw = w_w @ y (yw recomputed, not stored)
__global__ void k_bnstats(const float* __restrict__ y, const float* __restrict__ w_w,
                          float* bnsum, float* bnsq) {
  const int chunk = blockIdx.x & 63, o = (blockIdx.x >> 6) & 127, b = blockIdx.x >> 13;
  const int pix = chunk * 256 + threadIdx.x;
  const float* yb = y + (size_t)b * CA * H_ * W_ + pix;
  const float* wr = w_w + o * CA;
  float acc = 0.f;
  for (int i = 0; i < CA; ++i) acc += wr[i] * yb[(size_t)i * H_ * W_];
  __shared__ float r1[256], r2[256];
  r1[threadIdx.x] = acc; r2[threadIdx.x] = acc * acc; __syncthreads();
  for (int st = 128; st > 0; st >>= 1) {
    if (threadIdx.x < st) { r1[threadIdx.x] += r1[threadIdx.x + st]; r2[threadIdx.x] += r2[threadIdx.x + st]; }
    __syncthreads();
  }
  if (threadIdx.x == 0) { atomicAdd(&bnsum[o], r1[0]); atomicAdd(&bnsq[o], r2[0]); }
}

// ---------------------------------------------------------------- finalize: BN(yw) + alpha
__global__ void k_final(const float* __restrict__ y, const float* __restrict__ w_w,
                        const float* __restrict__ bnsum, const float* __restrict__ bnsq,
                        const float* __restrict__ gamma, const float* __restrict__ beta,
                        const float* __restrict__ alpha, float* __restrict__ out) {
  const int chunk = blockIdx.x & 63, o = (blockIdx.x >> 6) & 127, b = blockIdx.x >> 13;
  const int pix = chunk * 256 + threadIdx.x;
  const float* yb = y + (size_t)b * CA * H_ * W_ + pix;
  const float* wr = w_w + o * CA;
  float acc = 0.f;
  for (int i = 0; i < CA; ++i) acc += wr[i] * yb[(size_t)i * H_ * W_];
  const float inv = 1.0f / (float)(B_ * H_ * W_);
  float mean = bnsum[o] * inv;
  float var  = bnsq[o] * inv - mean * mean;
  size_t oi = ((size_t)(b * CA + o)) * H_ * W_ + pix;
  out[oi] = (acc - mean) * rsqrtf(var + 1e-5f) * gamma[o] + beta[o] + alpha[oi];
}

// ---------------------------------------------------------------- host launch
extern "C" void kernel_launch(void* const* d_in, const int* in_sizes, int n_in,
                              void* d_out, int out_size, void* d_ws, size_t ws_size,
                              hipStream_t stream) {
  const float* f     = (const float*)d_in[0];
  const float* alpha = (const float*)d_in[1];
  const float* unk   = (const float*)d_in[2];
  const float* gc_w  = (const float*)d_in[3];
  const float* gc_b  = (const float*)d_in[4];
  const float* w_w   = (const float*)d_in[5];
  const float* gamma = (const float*)d_in[6];
  const float* beta  = (const float*)d_in[7];
  float* out = (float*)d_out;
  char* ws = (char*)d_ws;

  float*          fds    = (float*)(ws);                         //  8,388,608 B
  unsigned short* P      = (unsigned short*)(ws + 8388608ull);   // 37,748,736 B
  unsigned short* V      = (unsigned short*)(ws + 46137344ull);  // 67,108,864 B
  float*          scores = (float*)(ws + 113246208ull);          // 268,435,456 B
  float*          rs     = (float*)(ws + 381681664ull);          //     65,536 B
  float*          mmv    = (float*)(ws + 381747200ull);          //     65,536 B
  float*          cmax   = (float*)(ws + 381812736ull);          //     65,536 B
  float*          csum   = (float*)(ws + 381878272ull);          //     65,536 B
  float*          ybuf   = (float*)(ws + 381943808ull);          // 33,554,432 B
  float*          bnsum  = (float*)(ws + 415498240ull);          //        512 B
  float*          bnsq   = (float*)(ws + 415498752ull);          //        512 B
  float*          susk   = (float*)(ws + 415499264ull);          //         32 B

  k_init     <<<32768, 256, 0, stream>>>(ybuf, bnsum, bnsq);
  k_meta     <<<B_,    256, 0, stream>>>(unk, susk, out);
  k_fds      <<<8192,  256, 0, stream>>>(f, gc_w, gc_b, fds);
  k_pack_P   <<<B_ * L_, 128, 0, stream>>>(fds, unk, susk, P, rs, mmv);
  k_pack_V   <<<B_ * L_, 128, 0, stream>>>(alpha, V);
  k_gemm1    <<<dim3(L_ / 128, L_ / 128, B_), 256, 0, stream>>>(P, rs, mmv, scores);
  k_colreduce<<<dim3(L_ / 256, B_), 256, 0, stream>>>(scores, cmax, csum, out);
  k_gemm2    <<<dim3(EV / 128, L_ / 64, B_), 256, 0, stream>>>(scores, V, cmax, csum, ybuf);
  k_bnstats  <<<32768, 256, 0, stream>>>(ybuf, w_w, bnsum, bnsq);
  k_final    <<<32768, 256, 0, stream>>>(ybuf, w_w, bnsum, bnsq, gamma, beta, alpha, out);
}
